// HGNN_85727547228783
// MI455X (gfx1250) — matrix operand order repaired and checked by reading, not compiled, not run on hardware
//
#include <hip/hip_runtime.h>
#include <hip/hip_bf16.h>

typedef __attribute__((ext_vector_type(16))) __bf16 v16bf;
typedef __attribute__((ext_vector_type(8)))  float  v8f;

#define N_NODES  100000
#define N_EDGES  1600000
#define F_IN     128
#define H_DIM    64
#define C_OUT    8
#define N_TILES  6250          // N_NODES / 16
#define GEMM_GRID 782          // ceil(N_TILES / 8 waves)

__device__ __forceinline__ void atomAddF(float* p, float v) {
  __hip_atomic_fetch_add(p, v, __ATOMIC_RELAXED, __HIP_MEMORY_SCOPE_AGENT);
}

// ---------------------------------------------------------------------------
// Stage one 16-wide bf16 A-fragment for v_wmma_f32_16x16x32_bf16.
// Per ISA layout: lane<16 holds row=lane, K = {off..off+7, off+16..off+23}
// where off = kc*32 (+8 for lanes 16..31). Caller passes off already adjusted.
// ---------------------------------------------------------------------------
__device__ __forceinline__ v16bf stage_a(const float* __restrict__ xr, int off) {
  float4 p0 = *(const float4*)(xr + off);
  float4 p1 = *(const float4*)(xr + off + 4);
  float4 q0 = *(const float4*)(xr + off + 16);
  float4 q1 = *(const float4*)(xr + off + 20);
  v16bf a;
  a[0]  = (__bf16)p0.x; a[1]  = (__bf16)p0.y; a[2]  = (__bf16)p0.z; a[3]  = (__bf16)p0.w;
  a[4]  = (__bf16)p1.x; a[5]  = (__bf16)p1.y; a[6]  = (__bf16)p1.z; a[7]  = (__bf16)p1.w;
  a[8]  = (__bf16)q0.x; a[9]  = (__bf16)q0.y; a[10] = (__bf16)q0.z; a[11] = (__bf16)q0.w;
  a[12] = (__bf16)q1.x; a[13] = (__bf16)q1.y; a[14] = (__bf16)q1.z; a[15] = (__bf16)q1.w;
  return a;
}

// ---------------------------------------------------------------------------
// Convert weights to bf16, pre-swizzled into per-lane WMMA B layout:
// tile(t,kc,c): lane l gets 16 contiguous bf16 = column n=c*16+(l&15),
//               K = kc*32 + (l>>4)*8 + (j&7) + (j>>3)*16   (j = 0..15)
// ---------------------------------------------------------------------------
__global__ void prep_weights(const float* __restrict__ W_in,
                             const float* __restrict__ W_conv,
                             __bf16* __restrict__ wbin,
                             __bf16* __restrict__ wbconv) {
  int tid = blockIdx.x * 256 + threadIdx.x;   // 49152 total
  if (tid < 32768) {
    int j  = tid & 15;
    int l  = (tid >> 4) & 31;
    int c  = (tid >> 9) & 3;
    int kc = (tid >> 11) & 3;
    int t  = tid >> 13;
    int k  = kc * 32 + (l >> 4) * 8 + (j & 7) + (j >> 3) * 16;
    int n  = c * 16 + (l & 15);
    wbin[tid] = (__bf16)W_in[(t * F_IN + k) * H_DIM + n];
  } else {
    int u  = tid - 32768;
    int j  = u & 15;
    int l  = (u >> 4) & 31;
    int c  = (u >> 9) & 3;
    int kc = (u >> 11) & 1;
    int m  = u >> 12;
    int k  = kc * 32 + (l >> 4) * 8 + (j & 7) + (j >> 3) * 16;
    int n  = c * 16 + (l & 15);
    wbconv[u] = (__bf16)W_conv[(m * H_DIM + k) * H_DIM + n];
  }
}

// ---------------------------------------------------------------------------
// Degree / symmetric norm (self-loop included: init to 1.0)
// ---------------------------------------------------------------------------
__global__ void deg_init(float* d0, float* d1) {
  int i = blockIdx.x * 256 + threadIdx.x;
  if (i < N_NODES) { d0[i] = 1.0f; d1[i] = 1.0f; }
}
__global__ void deg_count(const int* __restrict__ dst, float* __restrict__ d) {
  int e = blockIdx.x * 256 + threadIdx.x;
  if (e < N_EDGES) atomAddF(&d[dst[e]], 1.0f);
}
__global__ void deg_fin(float* d0, float* d1) {
  int i = blockIdx.x * 256 + threadIdx.x;
  if (i < N_NODES) { d0[i] = rsqrtf(d0[i]); d1[i] = rsqrtf(d1[i]); }
}

// ---------------------------------------------------------------------------
// Input projection: hid[n,:] = x[n,:] @ W_in[type[n]] + b_in[type[n]]
// One wave = one 16-node tile. Type loop kept rolled and each type's result
// is conditionally stored straight to hid (every row matches exactly one
// type), so no cross-iteration accumulator survives the loop -> low VGPR
// pressure -> high occupancy for this HBM-streaming kernel.
// ---------------------------------------------------------------------------
__global__ __launch_bounds__(256) void input_proj(
    const float* __restrict__ x, const int* __restrict__ types,
    const __bf16* __restrict__ wb, const float* __restrict__ bin,
    float* __restrict__ hid) {
  int lane = threadIdx.x & 31, wave = threadIdx.x >> 5;
  int tile = blockIdx.x * 8 + wave;
  if (tile >= N_TILES) return;
  int base = tile * 16;
  int r = lane & 15, half = lane >> 4;
  const float* xr = x + (size_t)(base + r) * F_IN;

  v16bf a4[4];
#pragma unroll
  for (int kc = 0; kc < 4; ++kc) a4[kc] = stage_a(xr, kc * 32 + half * 8);

  int ty[8];
#pragma unroll
  for (int v = 0; v < 8; ++v) ty[v] = types[base + v + 8 * half];

  const v8f zero = {0.f, 0.f, 0.f, 0.f, 0.f, 0.f, 0.f, 0.f};

#pragma unroll 1
  for (int t = 0; t < 4; ++t) {
    v8f acc[4] = {zero, zero, zero, zero};
#pragma unroll
    for (int kc = 0; kc < 4; ++kc) {
#pragma unroll
      for (int c = 0; c < 4; ++c) {
        v16bf b = *(const v16bf*)(wb + (size_t)(((t * 4 + kc) * 4 + c) * 32 + lane) * 16);
        acc[c] = __builtin_amdgcn_wmma_f32_16x16x32_bf16(
            false, a4[kc], false, b, (short)0, acc[c], false, false);
      }
    }
#pragma unroll
    for (int c = 0; c < 4; ++c) {
      float bias = bin[t * H_DIM + c * 16 + r];
#pragma unroll
      for (int v = 0; v < 8; ++v)
        if (ty[v] == t)
          hid[(size_t)(base + v + 8 * half) * H_DIM + c * 16 + r] = acc[c][v] + bias;
    }
  }
}

// ---------------------------------------------------------------------------
// Conv projection: xw = X @ W  (64x64, bf16 WMMA, no bias here)
// ---------------------------------------------------------------------------
__global__ __launch_bounds__(256) void conv_proj(
    const float* __restrict__ X, const __bf16* __restrict__ wb,
    float* __restrict__ xw) {
  int lane = threadIdx.x & 31, wave = threadIdx.x >> 5;
  int tile = blockIdx.x * 8 + wave;
  if (tile >= N_TILES) return;
  int base = tile * 16;
  int r = lane & 15, half = lane >> 4;
  const float* xr = X + (size_t)(base + r) * H_DIM;

  v16bf a2[2];
#pragma unroll
  for (int kc = 0; kc < 2; ++kc) a2[kc] = stage_a(xr, kc * 32 + half * 8);

  const v8f zero = {0.f, 0.f, 0.f, 0.f, 0.f, 0.f, 0.f, 0.f};
  v8f acc[4] = {zero, zero, zero, zero};
#pragma unroll
  for (int kc = 0; kc < 2; ++kc) {
#pragma unroll
    for (int c = 0; c < 4; ++c) {
      v16bf b = *(const v16bf*)(wb + (size_t)(((kc * 4 + c) * 32) + lane) * 16);
      acc[c] = __builtin_amdgcn_wmma_f32_16x16x32_bf16(
          false, a2[kc], false, b, (short)0, acc[c], false, false);
    }
  }
#pragma unroll
  for (int c = 0; c < 4; ++c)
#pragma unroll
    for (int v = 0; v < 8; ++v)
      xw[(size_t)(base + v + 8 * half) * H_DIM + c * 16 + r] = acc[c][v];
}

// acc[n,h] = xw[n,h] * dinv[n]^2   (self-loop term; also initializes acc)
__global__ void self_init(const float* __restrict__ xw,
                          const float* __restrict__ dinv,
                          float* __restrict__ acc) {
  int i = blockIdx.x * 256 + threadIdx.x;      // N_NODES*64 = 6.4M exactly
  float di = dinv[i >> 6];
  acc[i] = xw[i] * di * di;
}

// acc[dst,:] += xw[src,:] * dinv[src]*dinv[dst]; one thread per (edge, 4 floats)
__global__ void scatter_k(const float* __restrict__ xw,
                          const int* __restrict__ src,
                          const int* __restrict__ dst,
                          const float* __restrict__ dinv,
                          float* __restrict__ acc) {
  long long tid = (long long)blockIdx.x * 256 + threadIdx.x;  // E*16 = 25.6M exactly
  int e = (int)(tid >> 4);
  int q = ((int)tid & 15) * 4;
  int s = src[e], d = dst[e];
  float nrm = dinv[s] * dinv[d];
  float4 v = *(const float4*)(xw + (size_t)s * H_DIM + q);
  float* ap = acc + (size_t)d * H_DIM + q;
  atomAddF(ap + 0, v.x * nrm);
  atomAddF(ap + 1, v.y * nrm);
  atomAddF(ap + 2, v.z * nrm);
  atomAddF(ap + 3, v.w * nrm);
}

// out[i] = max(a[i]+ba[h], b[i]+bb[h])
__global__ void combine_max(const float* __restrict__ a, const float* __restrict__ b,
                            const float* __restrict__ ba, const float* __restrict__ bb,
                            float* __restrict__ out) {
  int i = blockIdx.x * 256 + threadIdx.x;      // 6.4M exactly
  int h = i & (H_DIM - 1);
  out[i] = fmaxf(a[i] + ba[h], b[i] + bb[h]);
}

// max-combine + LayerNorm + exact-erf GELU + (@W3 + b3). softmax([N,1]) == 1.
__global__ __launch_bounds__(256) void final_k(
    const float* __restrict__ a0, const float* __restrict__ a1,
    const float* __restrict__ bA, const float* __restrict__ bB,
    const float* __restrict__ W3, const float* __restrict__ b3,
    float* __restrict__ out) {
  int n = blockIdx.x * 256 + threadIdx.x;
  if (n >= N_NODES) return;
  const size_t ro = (size_t)n * H_DIM;
  float v[H_DIM];
  float mu = 0.f;
#pragma unroll
  for (int h = 0; h < H_DIM; ++h) {
    float t = fmaxf(a0[ro + h] + bA[h], a1[ro + h] + bB[h]);
    v[h] = t;
    mu += t;
  }
  mu *= (1.0f / H_DIM);
  float var = 0.f;
#pragma unroll
  for (int h = 0; h < H_DIM; ++h) { float d = v[h] - mu; var += d * d; }
  var *= (1.0f / H_DIM);
  float rs = rsqrtf(var + 1e-5f);
#pragma unroll
  for (int h = 0; h < H_DIM; ++h) {
    float t = (v[h] - mu) * rs;
    v[h] = 0.5f * t * (1.0f + erff(t * 0.70710678118654752440f));
  }
#pragma unroll
  for (int c = 0; c < C_OUT; ++c) {
    float s = b3[c];
#pragma unroll
    for (int h = 0; h < H_DIM; ++h) s += v[h] * W3[h * C_OUT + c];
    out[(size_t)n * C_OUT + c] = s;
  }
}

extern "C" void kernel_launch(void* const* d_in, const int* in_sizes, int n_in,
                              void* d_out, int out_size, void* d_ws, size_t ws_size,
                              hipStream_t stream) {
  const float* x      = (const float*)d_in[0];
  const int*   types  = (const int*)d_in[1];
  const int*   e0     = (const int*)d_in[2];   // [2,E] flat: src then dst
  const int*   e1     = (const int*)d_in[3];
  const float* W_in   = (const float*)d_in[4];
  const float* b_in   = (const float*)d_in[5];
  const float* W_conv = (const float*)d_in[6];
  const float* b_conv = (const float*)d_in[7];
  const float* W3     = (const float*)d_in[12];
  const float* b3     = (const float*)d_in[13];
  float* out = (float*)d_out;

  char* ws = (char*)d_ws;
  __bf16* wbin   = (__bf16*)(ws + 0);            //  64 KB swizzled W_in
  __bf16* wbconv = (__bf16*)(ws + 65536);        //  32 KB swizzled W_conv
  float*  dinv0  = (float*)(ws + 98304);         //  N floats
  float*  dinv1  = (float*)(ws + 498432);        //  N floats
  float*  hid    = (float*)(ws + 898560);        //  N*H f32 (states[0])
  float*  s1     = (float*)(ws + 26498560);      //  N*H f32 (states[1])
  float*  xw     = (float*)(ws + 52098560);      //  N*H f32 scratch
  float*  acc0   = (float*)(ws + 77698560);      //  N*H f32
  float*  acc1   = (float*)(ws + 103298560);     //  N*H f32  (end ~128.9 MB)

  prep_weights<<<192, 256, 0, stream>>>(W_in, W_conv, wbin, wbconv);
  deg_init<<<391, 256, 0, stream>>>(dinv0, dinv1);
  deg_count<<<6250, 256, 0, stream>>>(e0 + N_EDGES, dinv0);
  deg_count<<<6250, 256, 0, stream>>>(e1 + N_EDGES, dinv1);
  deg_fin<<<391, 256, 0, stream>>>(dinv0, dinv1);

  input_proj<<<GEMM_GRID, 256, 0, stream>>>(x, types, wbin, b_in, hid);

  // conv0: state0, edges0  -> acc0
  conv_proj<<<GEMM_GRID, 256, 0, stream>>>(hid, wbconv + 0 * 4096, xw);
  self_init<<<25000, 256, 0, stream>>>(xw, dinv0, acc0);
  scatter_k<<<100000, 256, 0, stream>>>(xw, e0, e0 + N_EDGES, dinv0, acc0);
  // conv1: state0, edges1  -> acc1
  conv_proj<<<GEMM_GRID, 256, 0, stream>>>(hid, wbconv + 1 * 4096, xw);
  self_init<<<25000, 256, 0, stream>>>(xw, dinv1, acc1);
  scatter_k<<<100000, 256, 0, stream>>>(xw, e1, e1 + N_EDGES, dinv1, acc1);
  // states[1] = max(acc0 + b_conv0, acc1 + b_conv1)
  combine_max<<<25000, 256, 0, stream>>>(acc0, acc1, b_conv + 0 * 64, b_conv + 1 * 64, s1);

  // conv2: state0, edges0  -> acc0
  conv_proj<<<GEMM_GRID, 256, 0, stream>>>(hid, wbconv + 2 * 4096, xw);
  self_init<<<25000, 256, 0, stream>>>(xw, dinv0, acc0);
  scatter_k<<<100000, 256, 0, stream>>>(xw, e0, e0 + N_EDGES, dinv0, acc0);
  // conv3: state1, edges0  -> acc1
  conv_proj<<<GEMM_GRID, 256, 0, stream>>>(s1, wbconv + 3 * 4096, xw);
  self_init<<<25000, 256, 0, stream>>>(xw, dinv0, acc1);
  scatter_k<<<100000, 256, 0, stream>>>(xw, e0, e0 + N_EDGES, dinv0, acc1);

  final_k<<<391, 256, 0, stream>>>(acc0, acc1, b_conv + 2 * 64, b_conv + 3 * 64,
                                   W3, b3, out);
}
